// T_RED_GNN_20993800142924
// MI455X (gfx1250) — matrix operand order repaired and checked by reading, not compiled, not run on hardware
//
#include <hip/hip_runtime.h>
#include <hip/hip_bf16.h>

typedef __attribute__((ext_vector_type(16))) __bf16 v16bf;
typedef __attribute__((ext_vector_type(8)))  float  v8f;

#define B_SZ     32
#define N_NODES  100000
#define E_EDGES  1000000
#define H_DIM    64
#define L_LAYERS 3
#define NRELP1   231      /* N_REL + 1 */
#define N_ENT    7128
#define TILES    (E_EDGES / 16)
#define WPB      8        /* waves (16-edge tiles) per 256-thread block */
#define EDGE_BLOCKS ((TILES + WPB - 1) / WPB)

__device__ __forceinline__ void ld8(const float* __restrict__ p, float* d) {
  float4 a = *(const float4*)(p);
  float4 b = *(const float4*)(p + 4);
  d[0] = a.x; d[1] = a.y; d[2] = a.z; d[3] = a.w;
  d[4] = b.x; d[5] = b.y; d[6] = b.z; d[7] = b.w;
}

__device__ __forceinline__ void atomic_add_f32(float* p, float v) {
  __hip_atomic_fetch_add(p, v, __ATOMIC_RELAXED, __HIP_MEMORY_SCOPE_AGENT);
}

__global__ void zero_kernel(float* p, size_t n) {
  size_t i = (size_t)blockIdx.x * blockDim.x + threadIdx.x;
  size_t stride = (size_t)gridDim.x * blockDim.x;
  for (; i < n; i += stride) p[i] = 0.f;
}

// Pack Wf/Wn/Wp to bf16 [3][64][64]; pad W1 to bf16 [L][16][192] (rows >=5 zero);
// pad W2 to f32 [L][16].
__global__ void prep_kernel(const float* __restrict__ Wp, const float* __restrict__ Wn,
                            const float* __restrict__ Wf, const float* __restrict__ W1,
                            const float* __restrict__ W2,
                            __bf16* __restrict__ wbf, __bf16* __restrict__ w1bf,
                            float* __restrict__ w2pad) {
  int t = blockIdx.x * blockDim.x + threadIdx.x;
  if (t < 3 * 64 * 64) {
    int w = t >> 12;
    int r = t & 4095;
    const float* W = (w == 0) ? Wf : ((w == 1) ? Wn : Wp);
    wbf[t] = (__bf16)W[r];
  }
  if (t < 3 * 16 * 192) {
    int l = t / (16 * 192);
    int r = t % (16 * 192);
    int n = r / 192, k = r % 192;
    w1bf[t] = (__bf16)((n < 5) ? W1[(l * 5 + n) * 192 + k] : 0.f);
  }
  if (t < 3 * 16) {
    int l = t >> 4, n = t & 15;
    w2pad[t] = (n < 5) ? W2[l * 5 + n] : 0.f;
  }
}

__global__ __launch_bounds__(256) void edge_kernel(
    const float* __restrict__ hidden, float* __restrict__ hidden_next,
    const float* __restrict__ rela_l, const float* __restrict__ time_embed,
    const __bf16* __restrict__ wbf, const __bf16* __restrict__ w1bf_l,
    const float* __restrict__ w2pad_l,
    const int* __restrict__ src_idx, const int* __restrict__ dst_idx,
    const int* __restrict__ rel_idx, const int* __restrict__ rel_time,
    const int* __restrict__ batch_idx, const int* __restrict__ query_rel) {
  // Stage all B-matrices in LDS (reused by every tile of every wave).
  __shared__ __attribute__((aligned(32))) __bf16 sWbf[3 * 64 * 64];  // 24.0 KB
  __shared__ __attribute__((aligned(32))) __bf16 sW1[16 * 192];      //  6.0 KB
  __shared__ float sW2[16];

  // Vectorized b128 fill: 24576 B -> 1536 uint4, 6144 B -> 384 uint4.
  {
    const uint4* gs = (const uint4*)wbf;
    uint4* ds = (uint4*)sWbf;
    for (int idx = threadIdx.x; idx < (3 * 64 * 64) / 8; idx += 256) ds[idx] = gs[idx];
    const uint4* gs1 = (const uint4*)w1bf_l;
    uint4* ds1 = (uint4*)sW1;
    for (int idx = threadIdx.x; idx < (16 * 192) / 8; idx += 256) ds1[idx] = gs1[idx];
    if (threadIdx.x < 16) sW2[threadIdx.x] = w2pad_l[threadIdx.x];
  }
  __syncthreads();

  const int wave = threadIdx.x >> 5;
  const int lane = threadIdx.x & 31;
  const int tile = blockIdx.x * WPB + wave;
  if (tile >= TILES) return;   // wave-uniform: EXEC stays all-ones for WMMA
  const int half = lane >> 4;  // K-half select for A/B fragments
  const int ml = lane & 15;    // A: row M; B/C: column N

  // Each lane fetches the edge scalars for row M = ml (halves duplicate; L0 hit).
  const int e   = tile * 16 + ml;
  const int src = src_idx[e];
  const int rel = rel_idx[e];
  const int rt  = rel_time[e];
  const int qr  = query_rel[batch_idx[e]];
  const int ta  = rt < 0 ? -rt : rt;

  const float* hr   = hidden     + (size_t)src * H_DIM;
  const float* rr   = rela_l     + (size_t)rel * H_DIM;
  const float* tr   = time_embed + (size_t)ta  * H_DIM;
  const float* qrow = rela_l     + (size_t)qr  * H_DIM;

  v8f accW[3][4];  // 3 weight branches x 4 N-tiles of the 16x64 result
  v8f accA;        // attention 16x16 (cols 0..4 valid)
#pragma unroll
  for (int w = 0; w < 3; ++w)
#pragma unroll
    for (int nt = 0; nt < 4; ++nt)
#pragma unroll
      for (int j = 0; j < 8; ++j) accW[w][nt][j] = 0.f;
#pragma unroll
  for (int j = 0; j < 8; ++j) accA[j] = 0.f;

#pragma unroll
  for (int ks = 0; ks < 2; ++ks) {  // K = ks*32 .. ks*32+31
    // 16-bit A fragment: lane (half,ml) holds row ml, K = kb..kb+15
    const int kb = ks * 32 + half * 16;
    float hb[16], rb[16], tb[16], qb[16];
    ld8(hr + kb, hb);       ld8(hr + kb + 8, hb + 8);
    ld8(rr + kb, rb);       ld8(rr + kb + 8, rb + 8);
    ld8(tr + kb, tb);       ld8(tr + kb + 8, tb + 8);
    ld8(qrow + kb, qb);     ld8(qrow + kb + 8, qb + 8);

    v16bf aH, aR, aQ, aE;
#pragma unroll
    for (int j = 0; j < 16; ++j) {
      aH[j] = (__bf16)hb[j];
      aR[j] = (__bf16)rb[j];
      aQ[j] = (__bf16)qb[j];
      aE[j] = (__bf16)(hb[j] + rb[j] + tb[j]);  // embed = hidden[src]+rel_e+t_emb
    }

    // transformed = embed @ {Wf,Wn,Wp}.T : B[k][n] = W[o][k], o = nt*16 + ml
#pragma unroll
    for (int w = 0; w < 3; ++w) {
#pragma unroll
      for (int nt = 0; nt < 4; ++nt) {
        const v16bf bfrag = *(const v16bf*)(sWbf + ((w * 64 + nt * 16 + ml) * 64 + ks * 32 + half * 16));
        accW[w][nt] = __builtin_amdgcn_wmma_f32_16x16x32_bf16(
            false, aE, false, bfrag, (short)0, accW[w][nt], false, false);
      }
    }
    // attention: attn_in = [hidden[src] | rel_e | rela[q_rel]] (192) @ W1.T (padded 16 cols)
    {
      const v16bf b0 = *(const v16bf*)(sW1 + (ml * 192 + 0 + ks * 32 + half * 16));
      accA = __builtin_amdgcn_wmma_f32_16x16x32_bf16(false, aH, false, b0, (short)0, accA, false, false);
      const v16bf b1 = *(const v16bf*)(sW1 + (ml * 192 + 64 + ks * 32 + half * 16));
      accA = __builtin_amdgcn_wmma_f32_16x16x32_bf16(false, aR, false, b1, (short)0, accA, false, false);
      const v16bf b2 = *(const v16bf*)(sW1 + (ml * 192 + 128 + ks * 32 + half * 16));
      accA = __builtin_amdgcn_wmma_f32_16x16x32_bf16(false, aQ, false, b2, (short)0, accA, false, false);
    }
  }

  // score[m] = sigmoid( sum_n relu(accA[m][n]) * W2[n] ); C layout: lane holds n=ml,
  // m = v + 8*half. Reduce over n with xor-shuffles (stay inside each 16-lane half).
  const float w2n = sW2[ml];
  float s[8];
#pragma unroll
  for (int v = 0; v < 8; ++v) {
    float c = accA[v];
    c = c > 0.f ? c : 0.f;
    c *= w2n;
    c += __shfl_xor(c, 1);
    c += __shfl_xor(c, 2);
    c += __shfl_xor(c, 4);
    c += __shfl_xor(c, 8);
    s[v] = 1.f / (1.f + __expf(-c));
  }

  // msg = score * transformed(selected branch); scatter-add into hidden_next[dst]
#pragma unroll
  for (int v = 0; v < 8; ++v) {
    const int m  = v + 8 * half;
    const int em = tile * 16 + m;
    const int rtm = rel_time[em];
    const int dm  = dst_idx[em];
    const int w = (rtm > 0) ? 0 : ((rtm == 0) ? 1 : 2);  // Wf / Wn / Wp
    float* outrow = hidden_next + (size_t)dm * H_DIM;
    const float sc = s[v];
#pragma unroll
    for (int nt = 0; nt < 4; ++nt) {
      const float val =
          ((w == 0) ? accW[0][nt][v] : (w == 1) ? accW[1][nt][v] : accW[2][nt][v]) * sc;
      atomic_add_f32(outrow + nt * 16 + ml, val);
    }
  }
}

__global__ void relu_swap_kernel(float* __restrict__ hidden, float* __restrict__ hidden_next,
                                 size_t n) {
  size_t i = (size_t)blockIdx.x * blockDim.x + threadIdx.x;
  size_t stride = (size_t)gridDim.x * blockDim.x;
  for (; i < n; i += stride) {
    float v = hidden_next[i];
    hidden[i] = v > 0.f ? v : 0.f;
    hidden_next[i] = 0.f;
  }
}

__global__ void out_kernel(const float* __restrict__ hidden, const float* __restrict__ Wc,
                           const float* __restrict__ bc, const int* __restrict__ out_batch,
                           const int* __restrict__ out_ent, float* __restrict__ out) {
  int n = blockIdx.x * blockDim.x + threadIdx.x;
  if (n >= N_NODES) return;
  float acc = bc[0];
  const float* hr = hidden + (size_t)n * H_DIM;
#pragma unroll
  for (int h = 0; h < H_DIM; ++h) acc += hr[h] * Wc[h];
  out[out_batch[n] * N_ENT + out_ent[n]] = acc;
}

extern "C" void kernel_launch(void* const* d_in, const int* in_sizes, int n_in,
                              void* d_out, int out_size, void* d_ws, size_t ws_size,
                              hipStream_t stream) {
  (void)in_sizes; (void)n_in; (void)out_size; (void)ws_size;
  const int* batch_idx = (const int*)d_in[0];
  const int* src_idx   = (const int*)d_in[1];
  const int* dst_idx   = (const int*)d_in[2];
  const int* rel_idx   = (const int*)d_in[3];
  const int* rel_time  = (const int*)d_in[4];
  const int* query_rel = (const int*)d_in[5];
  const int* out_batch = (const int*)d_in[6];
  const int* out_ent   = (const int*)d_in[7];
  const float* rela       = (const float*)d_in[8];
  const float* time_embed = (const float*)d_in[9];
  const float* Wp = (const float*)d_in[10];
  const float* Wn = (const float*)d_in[11];
  const float* Wf = (const float*)d_in[12];
  const float* W1 = (const float*)d_in[13];
  const float* W2 = (const float*)d_in[14];
  const float* Wc = (const float*)d_in[15];
  const float* bc = (const float*)d_in[16];
  float* out = (float*)d_out;

  // Workspace layout (all L2-resident: 2 x 25.6 MB + ~43 KB)
  float*  hidden      = (float*)d_ws;
  float*  hidden_next = hidden + (size_t)N_NODES * H_DIM;
  __bf16* wbf         = (__bf16*)(hidden_next + (size_t)N_NODES * H_DIM);
  __bf16* w1bf        = wbf + 3 * 64 * 64;
  float*  w2pad       = (float*)(w1bf + 3 * 16 * 192);

  zero_kernel<<<2048, 256, 0, stream>>>(hidden, (size_t)2 * N_NODES * H_DIM);
  zero_kernel<<<256, 256, 0, stream>>>(out, (size_t)B_SZ * N_ENT);
  prep_kernel<<<(3 * 64 * 64 + 255) / 256, 256, 0, stream>>>(Wp, Wn, Wf, W1, W2,
                                                             wbf, w1bf, w2pad);
  for (int i = 0; i < L_LAYERS; ++i) {
    edge_kernel<<<EDGE_BLOCKS, 256, 0, stream>>>(
        hidden, hidden_next, rela + (size_t)i * NRELP1 * H_DIM, time_embed,
        wbf, w1bf + (size_t)i * 16 * 192, w2pad + i * 16,
        src_idx, dst_idx, rel_idx, rel_time, batch_idx, query_rel);
    relu_swap_kernel<<<2048, 256, 0, stream>>>(hidden, hidden_next,
                                               (size_t)N_NODES * H_DIM);
  }
  out_kernel<<<(N_NODES + 255) / 256, 256, 0, stream>>>(hidden, Wc, bc, out_batch,
                                                        out_ent, out);
}
